// StyleGAN2ConvBlock_5480378269900
// MI455X (gfx1250) — matrix operand order repaired and checked by reading, not compiled, not run on hardware
//
#include <hip/hip_runtime.h>
#include <hip/hip_bf16.h>

// ---------- types for CDNA5 WMMA ----------
typedef __attribute__((ext_vector_type(16))) __bf16 bf16x16;
typedef __attribute__((ext_vector_type(8)))  __bf16 bf16x8;
typedef __attribute__((ext_vector_type(8)))  float  v8f;
// vector-of-4-int matching the async-to-LDS builtin's parameter type
typedef int v4i __attribute__((vector_size(16)));

// gfx1250 async global->LDS path (ASYNCcnt), guarded so either toolchain compiles
#if __has_builtin(__builtin_amdgcn_global_load_async_to_lds_b128) && \
    __has_builtin(__builtin_amdgcn_s_wait_asynccnt)
#define USE_ASYNC_LDS 1
#else
#define USE_ASYNC_LDS 0
#endif

// f32 -> bf16 (round-to-nearest-even), stored/handled as ushort bit pattern
__device__ __forceinline__ unsigned short f2bf(float f) {
    unsigned u = __float_as_uint(f);
    unsigned r = (u + 0x7FFFu + ((u >> 16) & 1u)) >> 16;
    return (unsigned short)r;
}
__device__ __forceinline__ float bf2f(unsigned short h) {
    return __uint_as_float(((unsigned)h) << 16);
}

// ---------- kernel 1: style = (cond @ cond_w^T) / sqrt(512) ----------
__global__ __launch_bounds__(128)
void style_kernel(const float* __restrict__ cond,
                  const float* __restrict__ w0,
                  const float* __restrict__ w1,
                  float* __restrict__ s0, float* __restrict__ s1)
{
    const int b = blockIdx.x;          // 0..7
    const int which = blockIdx.y;      // 0,1
    const int ci = threadIdx.x;        // 0..127
    const float* w = which ? w1 : w0;
    const float* c = cond + (size_t)b * 512;
    const float* wr = w + (size_t)ci * 512;
    float s = 0.f;
    #pragma unroll 4
    for (int j = 0; j < 512; ++j) s += c[j] * wr[j];
    (which ? s1 : s0)[b * 128 + ci] = s * 0.044194173824159216f; // 1/sqrt(512)
}

// ---------- kernel 2: modulate + demodulate, write bf16 weights ----------
// output layout: wmod[b][kc=ci/32][tap][co][ci%32]  (bf16)
template<int CIN, int COUT>
__global__ __launch_bounds__(128)
void modweight_kernel(const float* __restrict__ conv_w,  // [COUT][CIN][3][3]
                      const float* __restrict__ style,   // [B][CIN]
                      unsigned short* __restrict__ wmod)
{
    __shared__ float red[128];
    const int b = blockIdx.x, co = blockIdx.y, ci = threadIdx.x;
    const float s = style[b * CIN + ci];
    float w[9]; float ss = 0.f;
    #pragma unroll
    for (int t = 0; t < 9; ++t) {
        float v = conv_w[((size_t)co * CIN + ci) * 9 + t] * s;
        w[t] = v; ss += v * v;
    }
    red[ci] = ss; __syncthreads();
    for (int off = 64; off > 0; off >>= 1) {
        if (ci < off) red[ci] += red[ci + off];
        __syncthreads();
    }
    const float demod = rsqrtf(red[0] + 1e-8f);
    const int kc = ci >> 5, cil = ci & 31;
    #pragma unroll
    for (int t = 0; t < 9; ++t)
        wmod[((((size_t)b * (CIN / 32) + kc) * 9 + t) * COUT + co) * 32 + cil] =
            f2bf(w[t] * demod);
}

// ---------- kernel 3: modulated 3x3 conv via implicit-GEMM WMMA ----------
// Workgroup: 256 threads (8 waves). Tile: 64 co x 32 pixels (one row y, 32 x).
// Wave (wm,wn): 16 co x 16 pixels. K-loop over ci in chunks of 32 (bf16 wmma).
template<int CIN, int COUT, bool IN_BF16, bool FUSE_RGB>
__global__ __launch_bounds__(256)
void conv_mod_kernel(const void* __restrict__ xin,            // f32 or bf16 [B][CIN][256][256]
                     const unsigned short* __restrict__ wmod, // [b][kc][tap][co][32] bf16
                     const float* __restrict__ bias,          // [COUT]
                     const float* __restrict__ noise,         // [B][COUT][256][256]
                     const float* __restrict__ nscale,        // [COUT]
                     unsigned short* __restrict__ out_bf,     // bf16 out (conv0 path)
                     float* __restrict__ out_f,               // f32 out (conv1 path)
                     const float* __restrict__ rgb_w,         // [3][64]
                     const float* __restrict__ rgb_b,         // [3]
                     float* __restrict__ rgb_out)             // [B][3][256][256]
{
    constexpr int KC = CIN / 32;
    __shared__ unsigned short Ws[9 * 64 * 32];   // [tap][co_l][cil]  36 KB
    __shared__ unsigned short Xs[3 * 34 * 32];   // [row][col][cil]   6.4 KB
    __shared__ float Obuf[64 * 32];              // rgb fusion        8 KB

    const int tid  = threadIdx.x;
    const int lane = tid & 31;
    const int wave = tid >> 5;
    const int wm   = wave >> 1;        // 0..3 (co tile)
    const int wn   = wave & 1;         // 0..1 (pixel tile)
    const int m    = lane & 15;
    const int hi   = lane >> 4;        // lane 16..31 half

    const int b      = blockIdx.y;
    const int cobase = blockIdx.z * 64;
    const int tile   = blockIdx.x;     // 0..2047
    const int y      = tile >> 3;
    const int x0     = (tile & 7) << 5;

    const float*          xf = (const float*)xin;
    const unsigned short* xb = (const unsigned short*)xin;

    v8f acc = {};

    // contiguous k order (k = 16*hi + s), same for A and B (16-bit A 16x32, B 32x16)
    const int ka = hi ? 16 : 0;   // A: K 0..15 (lanes<16), 16..31 (lanes>=16)
    const int kb = hi ? 16 : 0;   // B: K 0..15 (lanes<16), 16..31 (lanes>=16)
    const int aRow = wm * 16 + m;
    const int nl   = wn * 16 + m;

    for (int kc = 0; kc < KC; ++kc) {
        __syncthreads();   // protect LDS reuse across chunks

        // ---- stage weights: Ws[tap][co_l][0..32) <- wmod[b][kc][tap][cobase+co_l][0..32)
        // Uses gfx1250 async global->LDS (no VGPR bounce, ASYNCcnt) when available.
        {
            const uint4* src = (const uint4*)(wmod + (size_t)(b * KC + kc) * 9 * COUT * 32);
            uint4* dst = (uint4*)Ws;
            for (int d = tid; d < 9 * 64 * 32 / 8; d += 256) {
                const int q    = d & 3;
                const int row  = d >> 2;          // tap*64 + co_l
                const int co_l = row & 63;
                const int tap  = row >> 6;
                const uint4* sp = src + (tap * COUT + cobase + co_l) * 4 + q;
#if USE_ASYNC_LDS
                __builtin_amdgcn_global_load_async_to_lds_b128(
                    (__attribute__((address_space(1))) v4i*)sp,
                    (__attribute__((address_space(3))) v4i*)(dst + d),
                    /*imm offset*/0, /*cpol*/0);
#else
                dst[d] = *sp;
#endif
            }
        }
        // ---- stage input halo patch (3 rows x 34 cols x 32 ci), bf16, layout [row][col][ci]
        // (manual: needs f32->bf16 convert + zero padding + channel transpose)
        for (int idx = tid; idx < 3 * 34 * 32; idx += 256) {
            const int ci  = idx & 31;
            const int rc  = idx >> 5;
            const int col = rc % 34;
            const int row = rc / 34;
            const int yy  = y + row - 1;
            const int xx  = x0 + col - 1;
            float v = 0.f;
            if ((unsigned)yy < 256u && (unsigned)xx < 256u) {
                const size_t g = ((size_t)(b * CIN + kc * 32 + ci) << 16) + (yy << 8) + xx;
                v = IN_BF16 ? bf2f(xb[g]) : xf[g];
            }
            Xs[idx] = f2bf(v);
        }
#if USE_ASYNC_LDS
        __builtin_amdgcn_s_wait_asynccnt(0);
#endif
        __syncthreads();

        // ---- 9 taps of v_wmma_f32_16x16x32_bf16
        #pragma unroll
        for (int tap = 0; tap < 9; ++tap) {
            const int ky = tap / 3, kx = tap % 3;
            bf16x16 A;
            *(bf16x8*)&A         = *(const bf16x8*)&Ws[(tap * 64 + aRow) * 32 + ka];
            *(((bf16x8*)&A) + 1) = *(const bf16x8*)&Ws[(tap * 64 + aRow) * 32 + ka + 8];
            bf16x16 Bm = *(const bf16x16*)&Xs[((ky * 34 + nl + kx) * 32) + kb];
            acc = __builtin_amdgcn_wmma_f32_16x16x32_bf16(
                false, A, false, Bm, (short)0, acc, false, false);
        }
    }

    // ---- epilogue: bias + noise; write bf16 (conv0) or f32 + rgb fusion (conv1)
    const int pix = x0 + wn * 16 + m;
    #pragma unroll
    for (int r = 0; r < 8; ++r) {
        const int co_l = wm * 16 + hi * 8 + r;   // D layout: VGPR r -> M=r (+8 for hi lanes)
        const int co   = cobase + co_l;
        const size_t g = ((size_t)(b * COUT + co) << 16) + (y << 8) + pix;
        const float v  = acc[r] + bias[co] + noise[g] * nscale[co];
        if constexpr (FUSE_RGB) {
            out_f[g] = v;
            Obuf[co_l * 32 + (wn * 16 + m)] = v;
        } else {
            out_bf[g] = f2bf(v);
        }
    }
    if constexpr (FUSE_RGB) {
        __syncthreads();
        if (tid < 96) {
            const int o = tid >> 5, p = tid & 31;
            float s = rgb_b[o];
            #pragma unroll 8
            for (int c = 0; c < 64; ++c) s += Obuf[c * 32 + p] * rgb_w[o * 64 + c];
            rgb_out[((size_t)(b * 3 + o) << 16) + (y << 8) + x0 + p] = s;
        }
    }
}

// ---------- host launch ----------
extern "C" void kernel_launch(void* const* d_in, const int* in_sizes, int n_in,
                              void* d_out, int out_size, void* d_ws, size_t ws_size,
                              hipStream_t stream) {
    (void)in_sizes; (void)n_in; (void)out_size; (void)ws_size;
    const float* inputs = (const float*)d_in[0];
    const float* cond   = (const float*)d_in[1];
    const float* cw0    = (const float*)d_in[2];
    const float* cw1    = (const float*)d_in[3];
    const float* convw0 = (const float*)d_in[4];
    const float* convb0 = (const float*)d_in[5];
    const float* convw1 = (const float*)d_in[6];
    const float* convb1 = (const float*)d_in[7];
    const float* ns0    = (const float*)d_in[8];
    const float* ns1    = (const float*)d_in[9];
    const float* noise0 = (const float*)d_in[10];
    const float* noise1 = (const float*)d_in[11];
    const float* rgbw   = (const float*)d_in[12];
    const float* rgbb   = (const float*)d_in[13];

    // workspace map (all offsets 256B aligned)
    char* ws = (char*)d_ws;
    float*          s0   = (float*)(ws + 0);                       //  4 KB
    float*          s1   = (float*)(ws + 4096);                    //  4 KB
    unsigned short* wm0  = (unsigned short*)(ws + 8192);           //  2.25 MB
    unsigned short* wm1  = (unsigned short*)(ws + 8192 + 2359296); //  1.125 MB
    unsigned short* out0 = (unsigned short*)(ws + 8192 + 2359296 + 1179648); // 128 MB bf16

    float* outp = (float*)d_out;                       // [8][64][256][256]
    float* rgbp = outp + (size_t)8 * 64 * 65536;       // [8][3][256][256]

    style_kernel<<<dim3(8, 2), 128, 0, stream>>>(cond, cw0, cw1, s0, s1);
    modweight_kernel<128, 128><<<dim3(8, 128), 128, 0, stream>>>(convw0, s0, wm0);
    modweight_kernel<128,  64><<<dim3(8,  64), 128, 0, stream>>>(convw1, s1, wm1);

    // block 0: 128 -> 128, f32 in, bf16 out to ws
    conv_mod_kernel<128, 128, false, false><<<dim3(2048, 8, 2), 256, 0, stream>>>(
        inputs, wm0, convb0, noise0, ns0, out0, nullptr, nullptr, nullptr, nullptr);

    // block 1: 128 -> 64, bf16 in, f32 out to d_out, fused 1x1 to-RGB
    conv_mod_kernel<128, 64, true, true><<<dim3(2048, 8, 1), 256, 0, stream>>>(
        out0, wm1, convb1, noise1, ns1, nullptr, outp, rgbw, rgbb, rgbp);
}
